// PhysicsMambaBlock_81862076662141
// MI455X (gfx1250) — matrix operand order repaired and checked by reading, not compiled, not run on hardware
//
#include <hip/hip_runtime.h>

// ---------------------------------------------------------------------------
// Mamba block on MI455X (gfx1250, wave32, WMMA).
// GEMMs: bf16 WMMA, fp32 accum; double-buffered LDS tiles fed by the
// Tensor Data Mover (tensor_load_to_lds, TENSORcnt), guard-free tiles.
// Conv / softplus / SSM scan / gating fused in fp32 VALU kernels.
// ---------------------------------------------------------------------------

typedef __bf16 bf16;
typedef __bf16 v16bf __attribute__((ext_vector_type(16)));
typedef float  v8f   __attribute__((ext_vector_type(8)));
typedef unsigned int u32x4 __attribute__((ext_vector_type(4)));
typedef unsigned int u32x8 __attribute__((ext_vector_type(8)));

#define NB     2
#define LSEQ   1024
#define DMODEL 1024
#define DINNER 2048
#define NSTATE 16
#define DCONV  4
#define NPAD   128   // padded N for the x-proj GEMM (live cols = 32)

// ---------------------------------------------------------------------------
__global__ __launch_bounds__(256) void cvt_f32_bf16(const float* __restrict__ in,
                                                    bf16* __restrict__ out, int n) {
  int i = blockIdx.x * 256 + threadIdx.x;
  if (i < n) out[i] = (bf16)in[i];
}

__global__ __launch_bounds__(256) void fill_zero_bf16(bf16* __restrict__ p, int n) {
  int i = blockIdx.x * 256 + threadIdx.x;
  if (i < n) p[i] = (bf16)0.0f;
}

// ---------------------------------------------------------------------------
// TDM: DMA one 2-D tile (row_dwords x nrows, row stride stride_dwords, 4-B
// elements) from global memory into LDS at lds_addr, with LDS row padding of
// 4 dwords after every 16 dwords (pad_interval=3 -> 64B, pad_amount=3 -> 16B),
// matching the 80-B padded LDS rows the WMMA fragment readers use.
// D# layout per CDNA5 ISA; groups 2/3 NULL (2-D).
// ---------------------------------------------------------------------------
__device__ __forceinline__ void tdm_load_2d(unsigned lds_addr, const void* gaddr,
                                            unsigned row_dwords, unsigned nrows,
                                            unsigned stride_dwords) {
  unsigned long long ga = (unsigned long long)gaddr;
  u32x4 g0;
  g0[0] = 1u;                                            // count=1 (valid user D#)
  g0[1] = lds_addr;                                      // LDS byte address
  g0[2] = (unsigned)(ga & 0xffffffffu);                  // global_addr[31:0]
  g0[3] = (unsigned)((ga >> 32) & 0x1ffffffu)            // global_addr[56:32]
          | 0x80000000u;                                 // type=2 ("image")
  u32x8 g1;
  g1[0] = (2u << 16)                                     // data_size = 4 bytes
        | (1u << 20)                                     // pad_enable
        | (3u << 22)                                     // pad_interval: 16 dwords
        | (3u << 25);                                    // pad_amount: 4 dwords
  g1[1] = (row_dwords & 0xffffu) << 16;                  // tensor_dim0[15:0]
  g1[2] = ((row_dwords >> 16) & 0xffffu)                 // tensor_dim0[31:16]
        | ((nrows & 0xffffu) << 16);                     // tensor_dim1[15:0]
  g1[3] = ((nrows >> 16) & 0xffffu)                      // tensor_dim1[31:16]
        | ((row_dwords & 0xffffu) << 16);                // tile_dim0
  g1[4] = (nrows & 0xffffu);                             // tile_dim1 (tile_dim2=0)
  g1[5] = stride_dwords;                                 // tensor_dim0_stride[31:0]
  g1[6] = 0u;                                            // stride hi / dim1_stride lo
  g1[7] = 0u;                                            // dim1_stride hi
  asm volatile("tensor_load_to_lds %0, %1" :: "s"(g0), "s"(g1) : "memory");
}

// ---------------------------------------------------------------------------
// C[M,N] = A[M,K] * B[N,K]^T  (A,B bf16 row-major, C fp32 row-major)
// Requirements: M % 64 == 0, N % 128 == 0, K % 32 == 0 (caller pads).
// Block tile 64(M) x 128(N), BK=32, 8 waves; wave = 1 A-frag x 4 B-frags
// -> 4 v_wmma_f32_16x16x32_bf16 per K-step, fragments preloaded into
// distinct VGPR ranges so ds_loads pipeline ahead of the WMMA chain.
// LDS double-buffered; tiles staged by the TDM (wave 0 issues, TENSORcnt
// drained via s_wait_tensorcnt, then workgroup barrier).
// ---------------------------------------------------------------------------
__global__ __launch_bounds__(256) void gemm_bf16_nt(const bf16* __restrict__ A,
                                                    const bf16* __restrict__ B,
                                                    float* __restrict__ C,
                                                    int M, int N, int K) {
  __shared__ alignas(16) bf16 As[2][64][40];    // 40-elem (80 B) rows: 64B data + 16B pad
  __shared__ alignas(16) bf16 Bs[2][128][40];

  const int tid  = threadIdx.x;
  const int wave = tid >> 5;
  const int lane = tid & 31;
  const int m0 = blockIdx.y * 64;
  const int n0 = blockIdx.x * 128;

  const int mi = wave & 3;           // M subtile (16 rows) within block
  const int nq = (wave >> 2) * 4;    // first of 4 N subtiles
  const int hi  = lane >> 4;
  const int l16 = lane & 15;

  v8f acc0 = {}, acc1 = {}, acc2 = {}, acc3 = {};

  const unsigned strideDw = (unsigned)(K >> 1);    // dwords per source row

  auto issue = [&](int buf, int k0) {
    if (wave == 0) {
      unsigned la = (unsigned)(unsigned long long)&As[buf][0][0];
      unsigned lb = (unsigned)(unsigned long long)&Bs[buf][0][0];
      tdm_load_2d(la, A + (size_t)m0 * K + k0, 16u, 64u, strideDw);
      tdm_load_2d(lb, B + (size_t)n0 * K + k0, 16u, 128u, strideDw);
    }
  };

  issue(0, 0);
  if (wave == 0) __builtin_amdgcn_s_wait_tensorcnt(0);
  __syncthreads();

  const int nk = K >> 5;
  for (int kb = 0; kb < nk; ++kb) {
    const int cur = kb & 1;
    if (kb + 1 < nk) issue(cur ^ 1, (kb + 1) * 32);

    // A fragment (16x32): lanes 0-15 rows M, K {0..7,16..23}; lanes 16-31 K {8..15,24..31}
    v16bf a;
    const int am    = mi * 16 + l16;
    const int kbase = hi * 8;
#pragma unroll
    for (int h = 0; h < 16; ++h) {
      const int k = (h < 8) ? (h + kbase) : (h + 8 + kbase);
      a[h] = As[cur][am][k];
    }
    // B fragments (32x16): lane%16 = N col; lanes 0-15 K {0..7,16..23}, lanes 16-31 K {8..15,24..31}.
    // All four fragments materialized in distinct registers before the WMMA
    // chain so the LDS loads overlap the matrix pipe.
    v16bf b0, b1, b2, b3;
    const int bn = nq * 16 + l16;
#pragma unroll
    for (int h = 0; h < 16; ++h) {
      const int k = (h < 8) ? (h + hi * 8) : (h + 8 + hi * 8);
      b0[h] = Bs[cur][bn][k];
      b1[h] = Bs[cur][bn + 16][k];
      b2[h] = Bs[cur][bn + 32][k];
      b3[h] = Bs[cur][bn + 48][k];
    }
    acc0 = __builtin_amdgcn_wmma_f32_16x16x32_bf16(false, a, false, b0, (short)0, acc0, false, false);
    acc1 = __builtin_amdgcn_wmma_f32_16x16x32_bf16(false, a, false, b1, (short)0, acc1, false, false);
    acc2 = __builtin_amdgcn_wmma_f32_16x16x32_bf16(false, a, false, b2, (short)0, acc2, false, false);
    acc3 = __builtin_amdgcn_wmma_f32_16x16x32_bf16(false, a, false, b3, (short)0, acc3, false, false);

    if (wave == 0) __builtin_amdgcn_s_wait_tensorcnt(0);
    __syncthreads();
  }

  // C/D layout: VGPR r -> row (r + 8*lanehalf); N = lane%16
#pragma unroll
  for (int t = 0; t < 4; ++t) {
    const v8f acc = (t == 0) ? acc0 : (t == 1) ? acc1 : (t == 2) ? acc2 : acc3;
    const int col = n0 + (nq + t) * 16 + l16;
#pragma unroll
    for (int r = 0; r < 8; ++r) {
      const int row = m0 + mi * 16 + r + hi * 8;
      C[(size_t)row * N + col] = acc[r];
    }
  }
}

// ---------------------------------------------------------------------------
// depthwise causal conv(4) + bias + SiLU on u (first DI cols of xz):
// emits fp32 (scan / D-skip) and bf16 (x-proj GEMM operand).
// ---------------------------------------------------------------------------
__global__ __launch_bounds__(256) void conv_silu(const float* __restrict__ xz,
                                                 const float* __restrict__ conv_w,
                                                 const float* __restrict__ conv_b,
                                                 float* __restrict__ uc,
                                                 bf16* __restrict__ ucbf) {
  const int idx = blockIdx.x * 256 + threadIdx.x;
  if (idx >= NB * LSEQ * DINNER) return;
  const int d = idx % DINNER;
  const int l = (idx / DINNER) % LSEQ;
  const int b = idx / (DINNER * LSEQ);

  float acc = conv_b[d];
#pragma unroll
  for (int j = 0; j < DCONV; ++j) {
    const int li = l - (DCONV - 1) + j;
    if (li >= 0)
      acc += xz[((size_t)(b * LSEQ + li)) * (2 * DINNER) + d] * conv_w[d * DCONV + j];
  }
  const float s = acc / (1.0f + __expf(-acc));
  const size_t o = (size_t)(b * LSEQ + l) * DINNER + d;
  uc[o]   = s;
  ucbf[o] = (bf16)s;
}

// ---------------------------------------------------------------------------
// Sequential SSM scan fused with softplus(dt), D-skip and SiLU(z) gating.
// One lane per (b,d) channel; 16 states in VGPRs; per-step B/C broadcast
// through LDS. grid = 16 blocks x 256 threads (B*DI = 4096 channels).
// ---------------------------------------------------------------------------
__global__ __launch_bounds__(256) void ssm_scan(const float* __restrict__ uconv,
                                                const float* __restrict__ xz,
                                                const float* __restrict__ BC,
                                                const float* __restrict__ dt_vals,
                                                const float* __restrict__ dt_w,
                                                const float* __restrict__ dt_b,
                                                const float* __restrict__ A_log,
                                                const float* __restrict__ Dvec,
                                                bf16* __restrict__ ybf) {
  __shared__ float sBC[32];
  const int b = blockIdx.x >> 3;
  const int d = (blockIdx.x & 7) * 256 + threadIdx.x;   // 0..DINNER-1

  float a[NSTATE], h[NSTATE];
#pragma unroll
  for (int n = 0; n < NSTATE; ++n) {
    a[n] = -__expf(A_log[d * NSTATE + n]);
    h[n] = 0.0f;
  }
  const float dtw = dt_w[d];
  const float dtb = dt_b[d];
  const float Dd  = Dvec[d];

  for (int l = 0; l < LSEQ; ++l) {
    const size_t row = (size_t)(b * LSEQ + l);
    if (threadIdx.x < 32) sBC[threadIdx.x] = BC[row * NPAD + threadIdx.x];
    __syncthreads();

    const float u   = uconv[row * DINNER + d];
    const float dtv = dt_vals[row];
    const float xdt = dtv * dtw + dtb;
    const float dt  = (xdt > 20.0f) ? xdt : log1pf(__expf(xdt));
    const float dtu = dt * u;

    float y = 0.0f;
#pragma unroll
    for (int n = 0; n < NSTATE; ++n) {
      h[n] = __expf(dt * a[n]) * h[n] + dtu * sBC[n];
      y   += h[n] * sBC[NSTATE + n];
    }

    const float z  = xz[row * (2 * DINNER) + DINNER + d];
    const float yg = (y + u * Dd) * (z / (1.0f + __expf(-z)));
    ybf[row * DINNER + d] = (bf16)yg;
    __syncthreads();
  }
}

// ---------------------------------------------------------------------------
// host-side launcher
// ---------------------------------------------------------------------------
extern "C" void kernel_launch(void* const* d_in, const int* in_sizes, int n_in,
                              void* d_out, int out_size, void* d_ws, size_t ws_size,
                              hipStream_t stream) {
  (void)in_sizes; (void)n_in; (void)out_size; (void)ws_size;

  const float* x       = (const float*)d_in[0];
  const float* dt_vals = (const float*)d_in[1];
  const float* W_in    = (const float*)d_in[2];
  const float* conv_w  = (const float*)d_in[3];
  const float* conv_b  = (const float*)d_in[4];
  const float* W_xproj = (const float*)d_in[5];
  const float* dt_w    = (const float*)d_in[6];
  const float* dt_b    = (const float*)d_in[7];
  const float* A_log   = (const float*)d_in[8];
  const float* Dv      = (const float*)d_in[9];
  const float* W_out   = (const float*)d_in[10];
  float* out = (float*)d_out;

  const int M = NB * LSEQ;                 // 2048 token rows
  char* ws = (char*)d_ws;
  size_t off = 0;
  auto alloc = [&](size_t bytes) -> char* {
    char* p = ws + off;
    off += (bytes + 255) & ~(size_t)255;
    return p;
  };

  bf16*  xbf    = (bf16*) alloc((size_t)M * DMODEL * 2);
  bf16*  winbf  = (bf16*) alloc((size_t)2 * DINNER * DMODEL * 2);
  bf16*  wxpbf  = (bf16*) alloc((size_t)NPAD * DINNER * 2);     // rows 32..127 zero
  bf16*  woutbf = (bf16*) alloc((size_t)DMODEL * DINNER * 2);
  float* xz     = (float*)alloc((size_t)M * 2 * DINNER * 4);
  float* uconv  = (float*)alloc((size_t)M * DINNER * 4);
  bf16*  ucbf   = (bf16*) alloc((size_t)M * DINNER * 2);
  float* BC     = (float*)alloc((size_t)M * NPAD * 4);
  bf16*  ybf    = (bf16*) alloc((size_t)M * DINNER * 2);

  const int DTR = 64;  // dead rows of W_xproj; only rows [DTR, DTR+32) are live

  // 1. bf16 conversions of GEMM operands (+ zero-pad x-proj weight)
  {
    int n = M * DMODEL;
    cvt_f32_bf16<<<(n + 255) / 256, 256, 0, stream>>>(x, xbf, n);
    n = 2 * DINNER * DMODEL;
    cvt_f32_bf16<<<(n + 255) / 256, 256, 0, stream>>>(W_in, winbf, n);
    n = 2 * NSTATE * DINNER;
    cvt_f32_bf16<<<(n + 255) / 256, 256, 0, stream>>>(W_xproj + (size_t)DTR * DINNER, wxpbf, n);
    int nz = (NPAD - 2 * NSTATE) * DINNER;
    fill_zero_bf16<<<(nz + 255) / 256, 256, 0, stream>>>(wxpbf + (size_t)2 * NSTATE * DINNER, nz);
    n = DMODEL * DINNER;
    cvt_f32_bf16<<<(n + 255) / 256, 256, 0, stream>>>(W_out, woutbf, n);
  }

  // 2. in-proj: xz[M, 2*DI] = x[M,DM] @ W_in[2*DI,DM]^T
  {
    dim3 grid((2 * DINNER) / 128, M / 64);
    gemm_bf16_nt<<<grid, 256, 0, stream>>>(xbf, winbf, xz, M, 2 * DINNER, DMODEL);
  }

  // 3. causal depthwise conv + SiLU on u half
  {
    int n = NB * LSEQ * DINNER;
    conv_silu<<<(n + 255) / 256, 256, 0, stream>>>(xz, conv_w, conv_b, uconv, ucbf);
  }

  // 4. x-proj (live B/C cols + zero pad): BC[M,128] = uconv[M,DI] @ Wxp_pad[128,DI]^T
  {
    dim3 grid(NPAD / 128, M / 64);
    gemm_bf16_nt<<<grid, 256, 0, stream>>>(ucbf, wxpbf, BC, M, NPAD, DINNER);
  }

  // 5. fused softplus + SSM scan + D-skip + SiLU gate -> ybf
  ssm_scan<<<16, 256, 0, stream>>>(uconv, xz, BC, dt_vals, dt_w, dt_b, A_log, Dv, ybf);

  // 6. out-proj: out[M,DM] = ybf[M,DI] @ W_out[DM,DI]^T
  {
    dim3 grid(DMODEL / 128, M / 64);
    gemm_bf16_nt<<<grid, 256, 0, stream>>>(ybf, woutbf, out, M, DMODEL, DINNER);
  }
}
